// ColBERTSim_11081015624292
// MI455X (gfx1250) — matrix operand order repaired and checked by reading, not compiled, not run on hardware
//
#include <hip/hip_runtime.h>

// ---------------------------------------------------------------------------
// ColBERT max-sim scoring on MI455X (gfx1250, wave32, WMMA).
//
//   cand_rep : [B=8,  NC=64, CL=32,  D=128] f32
//   ctxt_rep : [B=8,  NK=64, TL=128, D=128] f32
//   out      : [B, NC, NK] f32
//
// Compute-bound (AI ~800 FLOP/B; 42 MB working set is L2-resident).
// Core op: V_WMMA_F32_16X16X32_BF16 with a bf16 hi/lo split of the f32
// inputs (3 bf16 WMMAs ~= one exact-f32 K=32 step, vs 8 f32 16x16x4 WMMAs).
// ---------------------------------------------------------------------------

#define NEGC (-99999.0f)

typedef __attribute__((ext_vector_type(16))) __bf16 v16bf;
typedef __attribute__((ext_vector_type(8)))  float  v8f;

union Frag {            // 32 bytes = 8 VGPRs = one 16-bit WMMA A/B operand
  uint4 u[2];
  v16bf v;
};

__device__ __forceinline__ v8f bwmma(v16bf a, v16bf b, v8f c) {
  // D = A(16x32 bf16) x B(32x16 bf16) + C(16x16 f32)
  return __builtin_amdgcn_wmma_f32_16x16x32_bf16(
      /*neg_a=*/false, a, /*neg_b=*/false, b,
      /*c_mod=*/(short)0, c, /*reuse_a=*/false, /*reuse_b=*/false);
}

// ---------------------------------------------------------------------------
// Pass 1: split every f32 into bf16 hi + bf16 lo residual (x = hi + lo).
// Done once so the 64x-reused ctxt tensor is converted once, not per block.
// ---------------------------------------------------------------------------
__global__ void convert_split_kernel(const float* __restrict__ cand,
                                     const float* __restrict__ ctxt,
                                     __bf16* __restrict__ ch, __bf16* __restrict__ cl,
                                     __bf16* __restrict__ th, __bf16* __restrict__ tlo,
                                     int ncand, int nctxt) {
  const int tid = blockIdx.x * blockDim.x + threadIdx.x;
  const int stride = gridDim.x * blockDim.x;
  for (int i = tid; i < ncand; i += stride) {
    float x = cand[i];
    __bf16 h = (__bf16)x;            // RNE truncation
    ch[i] = h;
    cl[i] = (__bf16)(x - (float)h);  // residual
  }
  for (int i = tid; i < nctxt; i += stride) {
    float x = ctxt[i];
    __bf16 h = (__bf16)x;
    th[i] = h;
    tlo[i] = (__bf16)(x - (float)h);
  }
}

// ---------------------------------------------------------------------------
// Pass 2: one block per (b,q); 8 waves; wave w handles k = w, w+8, ... .
// No LDS, no barriers. A fragments live in registers across all 64 k's.
// ---------------------------------------------------------------------------
__global__ __launch_bounds__(256, 1)
void colbert_wmma_kernel(const __bf16* __restrict__ Ah, const __bf16* __restrict__ Al,
                         const __bf16* __restrict__ Bh, const __bf16* __restrict__ Bl,
                         const unsigned char* __restrict__ mask_cand,
                         const unsigned char* __restrict__ mask_ctxt,
                         float* __restrict__ out) {
  constexpr int NC = 64, NK = 64, CL = 32, TL = 128, D = 128;

  const int bq   = blockIdx.x;          // b*NC + q
  const int b    = bq >> 6;
  const int wave = threadIdx.x >> 5;
  const int lane = threadIdx.x & 31;
  const int hi16 = lane >> 4;           // 0: lanes 0-15, 1: lanes 16-31
  const int l16  = lane & 15;

  // -- candidate-token validity bitmasks for the C-fragment rows this lane
  //    owns (C/D layout: VGPR r -> M = r + 8*hi16 within a 16-row tile).
  const unsigned char* mc = mask_cand + (long)bq * CL;
  unsigned vm0 = 0, vm1 = 0;
#pragma unroll
  for (int r = 0; r < 8; ++r) {
    const int c = r + 8 * hi16;
    vm0 |= (mc[c]      ? 1u : 0u) << r;   // M-tile 0: c in [0,16)
    vm1 |= (mc[16 + c] ? 1u : 0u) << r;   // M-tile 1: c in [16,32)
  }

  // -- load A fragments (hi & lo) once; reused for all 64 k iterations.
  //    A 16x32 bf16 layout: lanes 0-15 hold row M=l16, K = {0..7, 16..23};
  //    lanes 16-31 hold K = {8..15, 24..31}.  -> two 16B chunks at +0,+16.
  Frag a_h[2][4], a_l[2][4];
  const long aRow = ((long)bq * CL + l16) * D;
#pragma unroll
  for (int m = 0; m < 2; ++m) {
#pragma unroll
    for (int dc = 0; dc < 4; ++dc) {
      const long off = aRow + (long)(16 * m) * D + 32 * dc + 8 * hi16;
      a_h[m][dc].u[0] = *(const uint4*)(Ah + off);
      a_h[m][dc].u[1] = *(const uint4*)(Ah + off + 16);
      a_l[m][dc].u[0] = *(const uint4*)(Al + off);
      a_l[m][dc].u[1] = *(const uint4*)(Al + off + 16);
    }
  }

  for (int k = wave; k < NK; k += 8) {
    // B 32x16 bf16 layout: lane holds column N=l16 (ctxt token t),
    // lanes 0-15 K = {0..7, 16..23}; lanes 16-31 K = {8..15, 24..31} -> +8*hi16.
    const long bRow = ((long)(b * NK + k) * TL + l16) * D + 8 * hi16;
    const unsigned char* mt = mask_ctxt + (long)(b * NK + k) * TL;

    // Prefetch the next k-tile this wave will touch (L2 warm-up hint).
    if (k + 8 < NK) {
      __builtin_prefetch(Bh + bRow + (long)8 * TL * D, 0, 1);
      __builtin_prefetch(Bl + bRow + (long)8 * TL * D, 0, 1);
    }

    float tsum = 0.f, msum = 0.f;

#pragma unroll 2
    for (int tc = 0; tc < 8; ++tc) {        // 8 column tiles of 16 ctxt tokens
      v8f acc0 = {};                        // scores rows c = 0..15
      v8f acc1 = {};                        // scores rows c = 16..31
#pragma unroll
      for (int dc = 0; dc < 4; ++dc) {      // K = 128 in 4 steps of 32
        Frag bh, bl;
        const long off = bRow + (long)(16 * tc) * D + 32 * dc;
        bh.u[0] = *(const uint4*)(Bh + off);
        bh.u[1] = *(const uint4*)(Bh + off + 16);
        bl.u[0] = *(const uint4*)(Bl + off);
        bl.u[1] = *(const uint4*)(Bl + off + 16);
        // hi*hi + hi*lo + lo*hi  ~= exact f32 product, f32 accumulate
        acc0 = bwmma(a_h[0][dc].v, bh.v, acc0);
        acc0 = bwmma(a_h[0][dc].v, bl.v, acc0);
        acc0 = bwmma(a_l[0][dc].v, bh.v, acc0);
        acc1 = bwmma(a_h[1][dc].v, bh.v, acc1);
        acc1 = bwmma(a_h[1][dc].v, bl.v, acc1);
        acc1 = bwmma(a_l[1][dc].v, bh.v, acc1);
      }

      // masked max over candidate rows held in this lane (8 per tile half)
      float rmax = NEGC;
#pragma unroll
      for (int r = 0; r < 8; ++r) {
        rmax = fmaxf(rmax, ((vm0 >> r) & 1u) ? acc0[r] : NEGC);
        rmax = fmaxf(rmax, ((vm1 >> r) & 1u) ? acc1[r] : NEGC);
      }
      // lanes n and n+16 hold the same ctxt token t -> combine halves
      rmax = fmaxf(rmax, __shfl_xor(rmax, 16, 32));

      // masked mean accumulation over ctxt tokens (t = 16*tc + l16)
      const float m = mt[16 * tc + l16] ? 1.f : 0.f;
      tsum += m * rmax;
      msum += m;
    }

    // Butterfly sum over all 32 lanes. Lane pairs (n, n+16) carry duplicate
    // values, so both numerator and denominator are doubled: ratio is exact.
#pragma unroll
    for (int off = 16; off >= 1; off >>= 1) {
      tsum += __shfl_xor(tsum, off, 32);
      msum += __shfl_xor(msum, off, 32);
    }
    if (lane == 0) out[(long)bq * NK + k] = tsum / msum;
  }
}

// ---------------------------------------------------------------------------
extern "C" void kernel_launch(void* const* d_in, const int* in_sizes, int n_in,
                              void* d_out, int out_size, void* d_ws, size_t ws_size,
                              hipStream_t stream) {
  const float*         cand  = (const float*)d_in[0];
  const float*         ctxt  = (const float*)d_in[1];
  const unsigned char* mcand = (const unsigned char*)d_in[2];  // jnp bool = 1 byte
  const unsigned char* mctxt = (const unsigned char*)d_in[3];
  float*               out   = (float*)d_out;

  const long NCAND = 8L * 64 * 32 * 128;    // 2,097,152 elements
  const long NCTXT = 8L * 64 * 128 * 128;   // 8,388,608 elements

  // Workspace layout (bf16): cand_hi | cand_lo | ctxt_hi | ctxt_lo = ~40 MB.
  __bf16* ch  = (__bf16*)d_ws;
  __bf16* cl  = ch + NCAND;
  __bf16* th  = cl + NCAND;
  __bf16* tlo = th + NCTXT;

  convert_split_kernel<<<2048, 256, 0, stream>>>(cand, ctxt, ch, cl, th, tlo,
                                                 (int)NCAND, (int)NCTXT);

  // one block per (b, q): 8 * 64 = 512 blocks, 8 waves each
  colbert_wmma_kernel<<<512, 256, 0, stream>>>(ch, cl, th, tlo, mcand, mctxt, out);
}
